// BitMarModel_56693568307298
// MI455X (gfx1250) — 16-bit and 8-bit matrix operand order repaired and checked by reading, compiled, not run on hardware
//
#include <hip/hip_runtime.h>
#include <stdint.h>

#define L   4
#define B   8
#define S   512
#define DIM 1024
#define H   16
#define FF  4096
#define HD  64
#define BS  (B*S)
#define EPS 1e-5f

typedef __attribute__((ext_vector_type(8)))  int       v8i;
typedef __attribute__((ext_vector_type(8)))  float     v8f;
typedef __attribute__((ext_vector_type(16))) _Float16  v16h;

static __device__ __forceinline__ v8i pack_v8i(int4 a, int4 b){
  union { int4 i[2]; v8i v; } u; u.i[0]=a; u.i[1]=b; return u.v;
}
static __device__ __forceinline__ v16h pack_v16h(int4 a, int4 b){
  union { int4 i[2]; v16h v; } u; u.i[0]=a; u.i[1]=b; return u.v;
}

static __device__ __forceinline__ void async_ld16(void* lds, const void* gptr){
  unsigned dst = (unsigned)(uintptr_t)lds;
  unsigned long long src = (unsigned long long)(uintptr_t)gptr;
  asm volatile("global_load_async_to_lds_b128 %0, %1, off"
               :: "v"(dst), "v"(src) : "memory");
}
static __device__ __forceinline__ void async_wait0(){
  asm volatile("s_wait_asynccnt 0" ::: "memory");
}

static __device__ __forceinline__ float blk_sum_f(float v){
  for(int o=16;o;o>>=1) v += __shfl_xor(v,o);
  __shared__ float sm[8];
  __syncthreads();
  if((threadIdx.x&31)==0) sm[threadIdx.x>>5]=v;
  __syncthreads();
  float t=0.f;
  #pragma unroll
  for(int i=0;i<8;++i) t += sm[i];
  return t;
}
static __device__ __forceinline__ int blk_sum_i(int v){
  for(int o=16;o;o>>=1) v += __shfl_xor(v,o);
  __shared__ int sm[8];
  __syncthreads();
  if((threadIdx.x&31)==0) sm[threadIdx.x>>5]=v;
  __syncthreads();
  int t=0;
  #pragma unroll
  for(int i=0;i<8;++i) t += sm[i];
  return t;
}

__global__ void k_init_wsum(float* wsum){ if(threadIdx.x<24) wsum[threadIdx.x]=0.f; }

__global__ void __launch_bounds__(256) k_wsum(const float* __restrict__ w, int n, float* slot){
  float s=0.f;
  for(int i=blockIdx.x*blockDim.x+threadIdx.x; i<n; i+=gridDim.x*blockDim.x) s += fabsf(w[i]);
  float t = blk_sum_f(s);
  if(threadIdx.x==0) atomicAdd(slot, t);
}

__global__ void __launch_bounds__(256) k_wquant(const float* __restrict__ w,
    int8_t* __restrict__ q, int* __restrict__ rowsum,
    const float* __restrict__ wsum, float* __restrict__ wscale, int rows, int cols){
  float sc = wsum[0] / (float)((size_t)rows*(size_t)cols);
  sc = fminf(fmaxf(sc, 1e-5f), 1000.f);
  int row = blockIdx.x;
  const float* wr = w + (size_t)row*cols;
  int8_t* qr = q + (size_t)row*cols;
  int s=0;
  for(int k=threadIdx.x;k<cols;k+=blockDim.x){
    float wn = fminf(fmaxf(wr[k]/sc, -10.f), 10.f);
    int t = (wn > 0.66666667f) ? 1 : ((wn < -0.66666667f) ? -1 : 0);
    qr[k] = (int8_t)t; s += t;
  }
  int tot = blk_sum_i(s);
  if(threadIdx.x==0) rowsum[row] = tot;
  if(row==0 && threadIdx.x==0) wscale[0] = sc;
}

__global__ void k_embed(const int* __restrict__ ids, const float* __restrict__ te,
                        const float* __restrict__ pe, float* __restrict__ x){
  for(size_t i=(size_t)blockIdx.x*blockDim.x+threadIdx.x; i<(size_t)BS*DIM;
      i += (size_t)gridDim.x*blockDim.x){
    int d = (int)(i % DIM); int bs = (int)(i / DIM); int s = bs % S;
    x[i] = te[(size_t)ids[bs]*DIM + d] + pe[(size_t)s*DIM + d];
  }
}

__global__ void __launch_bounds__(256) k_layernorm(const float* __restrict__ in1,
    const float* __restrict__ in2, const float* __restrict__ base,
    const float* __restrict__ w, const float* __restrict__ b, float* __restrict__ out){
  int row = blockIdx.x;
  const float* p1 = in1 + (size_t)row*DIM;
  const float* p2 = in2 ? in2 + (size_t)row*DIM : nullptr;
  float v[4]; float s=0.f;
  #pragma unroll
  for(int j=0;j<4;++j){
    int d = threadIdx.x + j*256;
    float t = p1[d]; if(p2) t += p2[d];
    v[j]=t; s+=t;
  }
  float mu = blk_sum_f(s) * (1.0f/DIM);
  float s2=0.f;
  #pragma unroll
  for(int j=0;j<4;++j){ float d0=v[j]-mu; s2 += d0*d0; }
  float var = blk_sum_f(s2) * (1.0f/DIM);
  float rs = rsqrtf(var + EPS);
  #pragma unroll
  for(int j=0;j<4;++j){
    int d = threadIdx.x + j*256;
    float o = (v[j]-mu)*rs*w[d] + b[d];
    if(base) o += base[(size_t)row*DIM + d];
    out[(size_t)row*DIM + d] = o;
  }
}

__global__ void k_mm_init(unsigned* mm){ if(threadIdx.x==0){ mm[0]=0xFFFFFFFFu; mm[1]=0u; } }

static __device__ __forceinline__ unsigned f2key(float v){
  unsigned bits = __float_as_uint(v);
  return (bits & 0x80000000u) ? ~bits : (bits | 0x80000000u);
}
static __device__ __forceinline__ float key2f(unsigned k){
  unsigned bits = (k & 0x80000000u) ? (k & 0x7FFFFFFFu) : ~k;
  return __uint_as_float(bits);
}

__global__ void __launch_bounds__(256) k_minmax(const void* __restrict__ src, size_t n,
                                                int is_half, unsigned* __restrict__ mm){
  unsigned lmin=0xFFFFFFFFu, lmax=0u;
  for(size_t i=(size_t)blockIdx.x*blockDim.x+threadIdx.x; i<n;
      i += (size_t)gridDim.x*blockDim.x){
    float v = is_half ? (float)((const _Float16*)src)[i] : ((const float*)src)[i];
    v = fminf(fmaxf(v,-1e6f),1e6f);
    unsigned k = f2key(v);
    lmin = k<lmin?k:lmin; lmax = k>lmax?k:lmax;
  }
  for(int o=16;o;o>>=1){
    unsigned t=__shfl_xor(lmin,o); lmin = t<lmin?t:lmin;
    t=__shfl_xor(lmax,o);          lmax = t>lmax?t:lmax;
  }
  if((threadIdx.x&31)==0){ atomicMin(mm+0,lmin); atomicMax(mm+1,lmax); }
}

__global__ void k_actparams(const unsigned* __restrict__ mm, float* __restrict__ ap){
  if(threadIdx.x==0){
    float mn=key2f(mm[0]), mx=key2f(mm[1]);
    float sc = fminf(fmaxf((mx-mn)/255.f, 1e-8f), 1000.f);
    float zp = fminf(fmaxf(rintf(-mn/sc), 0.f), 255.f);
    ap[0]=sc; ap[1]=zp;
  }
}

__global__ void __launch_bounds__(256) k_quant_act(const void* __restrict__ src, int is_half,
    size_t n, const float* __restrict__ ap, uint8_t* __restrict__ q){
  float sc=ap[0], zp=ap[1];
  for(size_t i=(size_t)blockIdx.x*blockDim.x+threadIdx.x; i<n;
      i += (size_t)gridDim.x*blockDim.x){
    float v = is_half ? (float)((const _Float16*)src)[i] : ((const float*)src)[i];
    v = fminf(fmaxf(v,-1e6f),1e6f);
    float t = fminf(fmaxf(rintf(v/sc + zp), 0.f), 255.f);
    q[i] = (uint8_t)t;
  }
}

template<int EPI>
__global__ void __launch_bounds__(256)
k_gemm_iu8(const uint8_t* __restrict__ A, const int8_t* __restrict__ Wq,
           const int* __restrict__ rowsum, const float* __restrict__ wscale,
           const float* __restrict__ ap, const float* __restrict__ bias,
           int K, int N, float* __restrict__ outf, _Float16* __restrict__ outh,
           const float* __restrict__ resid){
  __shared__ int8_t BT[2][32][80];
  const int tid  = threadIdx.x;
  const int lane = tid & 31;
  const int wv   = tid >> 5;
  const int half = lane >> 4;
  const int r    = lane & 15;
  const int m0   = blockIdx.y*128 + wv*16;
  const int n0   = blockIdx.x*32;
  const uint8_t* arow = A + (size_t)(m0 + r)*K + (half?16:0);
  const int hofs8 = half?16:0;
  const int srow  = tid >> 2;
  const int sboff = (tid & 3)*16;

  if(tid < 128)
    async_ld16(&BT[0][srow][sboff], Wq + (size_t)(n0+srow)*K + sboff);

  v8i acc0 = {}, acc1 = {};
  for(int k0=0;k0<K;k0+=64){
    int buf = (k0>>6)&1;
    async_wait0();
    __syncthreads();
    if(k0+64 < K && tid < 128)
      async_ld16(&BT[buf^1][srow][sboff], Wq + (size_t)(n0+srow)*K + (k0+64) + sboff);

    const uint2* ap2 = (const uint2*)(arow + k0);
    uint2 a0=ap2[0], a1=ap2[1], a2=ap2[4], a3=ap2[5];
    const int4* bp0 = (const int4*)&BT[buf][r][hofs8];
    const int4* bp1 = (const int4*)&BT[buf][16+r][hofs8];
    int4 c0=bp0[0], c1=bp0[2], d0=bp1[0], d1=bp1[2];
    v8i av; av[0]=(int)a0.x; av[1]=(int)a0.y; av[2]=(int)a1.x; av[3]=(int)a1.y;
            av[4]=(int)a2.x; av[5]=(int)a2.y; av[6]=(int)a3.x; av[7]=(int)a3.y;
    v8i bv0 = pack_v8i(c0,c1);
    v8i bv1 = pack_v8i(d0,d1);
    acc0 = __builtin_amdgcn_wmma_i32_16x16x64_iu8(false, av, true, bv0, acc0, false, false);
    acc1 = __builtin_amdgcn_wmma_i32_16x16x64_iu8(false, av, true, bv1, acc1, false, false);
  }
  float sa=ap[0], zp=ap[1], sw=wscale[0];
  const int nB0 = n0 + r, nB1 = n0 + 16 + r;
  float corr0 = zp * (float)rowsum[nB0];
  float corr1 = zp * (float)rowsum[nB1];
  float bb0 = bias[nB0], bb1 = bias[nB1];
  #pragma unroll
  for(int g=0; g<8; ++g){
    int mo = m0 + g + (half?8:0);
    float v0 = sa*sw*((float)acc0[g] - corr0) + bb0;
    float v1 = sa*sw*((float)acc1[g] - corr1) + bb1;
    size_t i0 = (size_t)mo*N + nB0;
    size_t i1 = (size_t)mo*N + nB1;
    if      (EPI==0){ outf[i0]=v0; outf[i1]=v1; }
    else if (EPI==1){ outh[i0]=(_Float16)v0; outh[i1]=(_Float16)v1; }
    else if (EPI==2){
      outh[i0]=(_Float16)(0.5f*v0*(1.0f + erff(v0*0.70710678f)));
      outh[i1]=(_Float16)(0.5f*v1*(1.0f + erff(v1*0.70710678f)));
    }else{
      outf[i0]=resid[i0]+v0; outf[i1]=resid[i1]+v1;
    }
  }
}

__global__ void __launch_bounds__(256)
k_attn(const _Float16* __restrict__ Q, const _Float16* __restrict__ Kt,
       const _Float16* __restrict__ V, _Float16* __restrict__ O){
  __shared__ _Float16 KT[64][72];
  __shared__ _Float16 VT[64][72];
  __shared__ _Float16 Pt[8][16][72];
  const int tid  = threadIdx.x;
  const int lane = tid & 31;
  const int wv   = tid >> 5;
  const int half = lane >> 4;
  const int r    = lane & 15;
  const int b    = blockIdx.x / H;
  const int hh   = blockIdx.x % H;
  const int m0   = blockIdx.y*128 + wv*16;
  const _Float16* qb = Q  + ((size_t)b*S)*DIM + hh*HD;
  const _Float16* kb = Kt + ((size_t)b*S)*DIM + hh*HD;
  const _Float16* vb = V  + ((size_t)b*S)*DIM + hh*HD;
  _Float16*       ob = O  + ((size_t)b*S)*DIM + hh*HD;

  const int bofs = half ? 8  : 0;
  const int hofs = half ? 16 : 0;

  const _Float16* qrow = qb + (size_t)(m0 + r)*DIM;
  const int4* qp0 = (const int4*)(qrow + hofs);
  v16h qa0 = pack_v16h(qp0[0], qp0[1]);
  const int4* qp1 = (const int4*)(qrow + 32 + hofs);
  v16h qa1 = pack_v16h(qp1[0], qp1[1]);

  float mrow[8], lrow[8], alpha[8];
  v8f acc[4] = {{},{},{},{}};
  #pragma unroll
  for(int g=0;g<8;++g){ mrow[g]=-3.0e38f; lrow[g]=0.f; }

  for(int j0=0;j0<S;j0+=64){
    __syncthreads();

    #pragma unroll
    for(int i=0;i<2;++i){
      int c   = tid + i*256;
      int key = c >> 3;
      int hc  = (c & 7)*8;
      async_ld16(&KT[key][hc], kb + (size_t)(j0+key)*DIM + hc);
    }
    #pragma unroll
    for(int i=0;i<2;++i){
      int c   = tid + i*256;
      int key = c >> 3;
      int hc  = (c & 7)*8;
      union { int4 q; _Float16 h[8]; } u;
      u.q = *(const int4*)(vb + (size_t)(j0+key)*DIM + hc);
      #pragma unroll
      for(int e=0;e<8;++e) VT[hc+e][key] = u.h[e];
    }
    async_wait0();
    __syncthreads();

    v8f sc[4];
    #pragma unroll
    for(int js=0;js<4;++js){
      v8f s = {};
      const _Float16* k0r = &KT[js*16 + r][hofs];
      v16h kf0 = pack_v16h(((const int4*)k0r)[0], ((const int4*)k0r)[1]);
      const _Float16* k1r = &KT[js*16 + r][32 + hofs];
      v16h kf1 = pack_v16h(((const int4*)k1r)[0], ((const int4*)k1r)[1]);
      s = __builtin_amdgcn_wmma_f32_16x16x32_f16(false,qa0,false,kf0,(short)0,s,false,false);
      s = __builtin_amdgcn_wmma_f32_16x16x32_f16(false,qa1,false,kf1,(short)0,s,false,false);
      sc[js] = s;
    }

    #pragma unroll
    for(int g=0;g<8;++g){
      float v0=sc[0][g]*0.125f, v1=sc[1][g]*0.125f;
      float v2=sc[2][g]*0.125f, v3=sc[3][g]*0.125f;
      float mx = fmaxf(fmaxf(v0,v1), fmaxf(v2,v3));
      for(int o=1;o<16;o<<=1) mx = fmaxf(mx, __shfl_xor(mx,o));
      float mn = fmaxf(mrow[g], mx);
      alpha[g] = expf(mrow[g]-mn);
      float p0=expf(v0-mn), p1=expf(v1-mn), p2=expf(v2-mn), p3=expf(v3-mn);
      float ps = (p0+p1)+(p2+p3);
      for(int o=1;o<16;o<<=1) ps += __shfl_xor(ps,o);
      lrow[g] = lrow[g]*alpha[g] + ps;
      mrow[g] = mn;
      int rr = g + (half?8:0);
      Pt[wv][rr][r]      = (_Float16)p0;
      Pt[wv][rr][16 + r] = (_Float16)p1;
      Pt[wv][rr][32 + r] = (_Float16)p2;
      Pt[wv][rr][48 + r] = (_Float16)p3;
    }
    __syncthreads();

    const int4* pp0 = (const int4*)(&Pt[wv][r][hofs]);
    v16h pa0 = pack_v16h(pp0[0], pp0[1]);
    const int4* pp1 = (const int4*)(&Pt[wv][r][32 + hofs]);
    v16h pa1 = pack_v16h(pp1[0], pp1[1]);

    #pragma unroll
    for(int t=0;t<4;++t){
      #pragma unroll
      for(int g=0;g<8;++g) acc[t][g] *= alpha[g];
    }
    #pragma unroll
    for(int t=0;t<4;++t){
      const _Float16* vt0 = &VT[t*16 + r][hofs];
      v16h vf0 = pack_v16h(((const int4*)vt0)[0], ((const int4*)vt0)[1]);
      const _Float16* vt1 = &VT[t*16 + r][32 + hofs];
      v16h vf1 = pack_v16h(((const int4*)vt1)[0], ((const int4*)vt1)[1]);
      acc[t] = __builtin_amdgcn_wmma_f32_16x16x32_f16(false, pa0, false, vf0, (short)0, acc[t], false, false);
      acc[t] = __builtin_amdgcn_wmma_f32_16x16x32_f16(false, pa1, false, vf1, (short)0, acc[t], false, false);
    }
  }

  #pragma unroll
  for(int g=0;g<8;++g){
    float inv = 1.0f / lrow[g];
    int so = m0 + g + (half?8:0);
    _Float16* orow = ob + (size_t)so*DIM;
    #pragma unroll
    for(int t=0;t<4;++t) orow[t*16 + r] = (_Float16)(acc[t][g]*inv);
  }
}

extern "C" void kernel_launch(void* const* d_in, const int* in_sizes, int n_in,
                              void* d_out, int out_size, void* d_ws, size_t ws_size,
                              hipStream_t stream){
  (void)in_sizes; (void)n_in; (void)out_size; (void)ws_size;
  const int*   ids  = (const int*)d_in[0];
  const float* te   = (const float*)d_in[1];
  const float* pe   = (const float*)d_in[2];
  const float* ln1w = (const float*)d_in[3];
  const float* ln1b = (const float*)d_in[4];
  const float* qw   = (const float*)d_in[5];
  const float* qb_  = (const float*)d_in[6];
  const float* kw   = (const float*)d_in[7];
  const float* kb_  = (const float*)d_in[8];
  const float* vw   = (const float*)d_in[9];
  const float* vb_  = (const float*)d_in[10];
  const float* ow   = (const float*)d_in[11];
  const float* ob_  = (const float*)d_in[12];
  const float* ln2w = (const float*)d_in[13];
  const float* ln2b = (const float*)d_in[14];
  const float* f1w  = (const float*)d_in[15];
  const float* f1b  = (const float*)d_in[16];
  const float* f2w  = (const float*)d_in[17];
  const float* f2b  = (const float*)d_in[18];
  const float* mlw  = (const float*)d_in[19];
  const float* mlb  = (const float*)d_in[20];
  const float* fnw  = (const float*)d_in[21];
  const float* fnb  = (const float*)d_in[22];

  char* wsp = (char*)d_ws;
  size_t off = 0;
  auto take = [&](size_t bytes)->char*{
    char* p = wsp + off; off += (bytes + 255) & ~(size_t)255; return p;
  };
  int8_t*   wqall = (int8_t*)take((size_t)L*12*1024*1024);
  int*      rsall = (int*)take((size_t)L*9216*sizeof(int));
  float*    wsum  = (float*)take(24*sizeof(float));
  float*    wscal = (float*)take(24*sizeof(float));
  unsigned* mm    = (unsigned*)take(2*sizeof(unsigned));
  float*    apar  = (float*)take(2*sizeof(float));
  float*    x     = (float*)take((size_t)BS*DIM*4);
  float*    hbuf  = (float*)take((size_t)BS*DIM*4);
  float*    ybuf  = (float*)take((size_t)BS*DIM*4);
  _Float16* g16   = (_Float16*)take((size_t)BS*FF*2);
  uint8_t*  a8    = (uint8_t*)take((size_t)BS*FF);
  _Float16* qh    = (_Float16*)take((size_t)BS*DIM*2);
  _Float16* kh    = (_Float16*)take((size_t)BS*DIM*2);
  _Float16* vh    = (_Float16*)take((size_t)BS*DIM*2);
  _Float16* ch    = (_Float16*)take((size_t)BS*DIM*2);

  k_init_wsum<<<1, 32, 0, stream>>>(wsum);

  for(int l=0;l<L;l++){
    struct WD { const float* src; int rows; int cols; size_t qoff; size_t roff; } tens[6] = {
      { qw  + (size_t)l*DIM*DIM, DIM, DIM, (size_t)l*12*1024*1024 + 0u,       (size_t)l*9216 + 0    },
      { kw  + (size_t)l*DIM*DIM, DIM, DIM, (size_t)l*12*1024*1024 + 1048576u, (size_t)l*9216 + 1024 },
      { vw  + (size_t)l*DIM*DIM, DIM, DIM, (size_t)l*12*1024*1024 + 2097152u, (size_t)l*9216 + 2048 },
      { ow  + (size_t)l*DIM*DIM, DIM, DIM, (size_t)l*12*1024*1024 + 3145728u, (size_t)l*9216 + 3072 },
      { f1w + (size_t)l*FF*DIM,  FF,  DIM, (size_t)l*12*1024*1024 + 4194304u, (size_t)l*9216 + 4096 },
      { f2w + (size_t)l*DIM*FF,  DIM, FF,  (size_t)l*12*1024*1024 + 8388608u, (size_t)l*9216 + 8192 },
    };
    for(int j=0;j<6;j++){
      int t = l*6 + j;
      int n = tens[j].rows*tens[j].cols;
      k_wsum<<<512, 256, 0, stream>>>(tens[j].src, n, wsum + t);
      k_wquant<<<tens[j].rows, 256, 0, stream>>>(tens[j].src, wqall + tens[j].qoff,
          rsall + tens[j].roff, wsum + t, wscal + t, tens[j].rows, tens[j].cols);
    }
  }

  k_embed<<<1024, 256, 0, stream>>>(ids, te, pe, x);

  auto quant = [&](const void* src, size_t n, int is_half){
    k_mm_init<<<1, 32, 0, stream>>>(mm);
    k_minmax<<<1024, 256, 0, stream>>>(src, n, is_half, mm);
    k_actparams<<<1, 32, 0, stream>>>(mm, apar);
    k_quant_act<<<1024, 256, 0, stream>>>(src, is_half, n, apar, a8);
  };

  const dim3 gg(DIM/32, BS/128);
  const dim3 g1(FF/32,  BS/128);

  for(int l=0;l<L;l++){
    int8_t* wq_l = wqall + (size_t)l*12*1024*1024;
    int*    rs_l = rsall + (size_t)l*9216;
    int     t0   = l*6;

    k_layernorm<<<BS, 256, 0, stream>>>(x, (const float*)nullptr, (const float*)nullptr,
        ln1w + (size_t)l*DIM, ln1b + (size_t)l*DIM, hbuf);
    quant(hbuf, (size_t)BS*DIM, 0);
    k_gemm_iu8<1><<<gg, 256, 0, stream>>>(a8, wq_l + 0,       rs_l + 0,    wscal+t0+0, apar,
        qb_ + (size_t)l*DIM, DIM, DIM, (float*)nullptr, qh, (const float*)nullptr);
    k_gemm_iu8<1><<<gg, 256, 0, stream>>>(a8, wq_l + 1048576, rs_l + 1024, wscal+t0+1, apar,
        kb_ + (size_t)l*DIM, DIM, DIM, (float*)nullptr, kh, (const float*)nullptr);
    k_gemm_iu8<1><<<gg, 256, 0, stream>>>(a8, wq_l + 2097152, rs_l + 2048, wscal+t0+2, apar,
        vb_ + (size_t)l*DIM, DIM, DIM, (float*)nullptr, vh, (const float*)nullptr);

    k_attn<<<dim3(B*H, S/128), 256, 0, stream>>>(qh, kh, vh, ch);

    quant(ch, (size_t)BS*DIM, 1);
    k_gemm_iu8<3><<<gg, 256, 0, stream>>>(a8, wq_l + 3145728, rs_l + 3072, wscal+t0+3, apar,
        ob_ + (size_t)l*DIM, DIM, DIM, x, (_Float16*)nullptr, x);

    k_layernorm<<<BS, 256, 0, stream>>>(x, (const float*)nullptr, (const float*)nullptr,
        ln2w + (size_t)l*DIM, ln2b + (size_t)l*DIM, ybuf);
    quant(ybuf, (size_t)BS*DIM, 0);
    k_gemm_iu8<2><<<g1, 256, 0, stream>>>(a8, wq_l + 4194304, rs_l + 4096, wscal+t0+4, apar,
        f1b + (size_t)l*FF, DIM, FF, (float*)nullptr, g16, (const float*)nullptr);
    quant(g16, (size_t)BS*FF, 1);
    k_gemm_iu8<0><<<gg, 256, 0, stream>>>(a8, wq_l + 8388608, rs_l + 8192, wscal+t0+5, apar,
        f2b + (size_t)l*DIM, FF, DIM, hbuf, (_Float16*)nullptr, (const float*)nullptr);
    k_layernorm<<<BS, 256, 0, stream>>>(hbuf, ybuf, x,
        mlw + (size_t)l*DIM, mlb + (size_t)l*DIM, x);
  }

  k_layernorm<<<BS, 256, 0, stream>>>(x, (const float*)nullptr, (const float*)nullptr,
      fnw, fnb, (float*)d_out);
}
